// DefectNetConv_53334903882517
// MI455X (gfx1250) — matrix operand order repaired and checked by reading, not compiled, not run on hardware
//
#include <hip/hip_runtime.h>
#include <hip/hip_bf16.h>
#include <math.h>

// Sizes fixed by the reference problem
#define FDIM 64     // F
#define GDIM 128    // G
#define KDIM 256    // 2F+G
#define MDIM 128    // 2F
#define EPSV 1e-5f
#define TROWS 32    // edges per block iteration (2 x 16-row WMMA tiles)

typedef __attribute__((ext_vector_type(16))) __bf16 v16bf;
typedef __attribute__((ext_vector_type(8)))  float  v8f;

union V16U { v16bf v; uint4 q[2]; };

__device__ __forceinline__ float sigmoidf_(float x) { return 1.0f / (1.0f + __expf(-x)); }
__device__ __forceinline__ float softplusf_(float x) {
    // numerically stable softplus
    return fmaxf(x, 0.0f) + log1pf(__expf(-fabsf(x)));
}

// ---------------------------------------------------------------------------
// Utility kernels
// ---------------------------------------------------------------------------
__global__ void zero_kernel(float* __restrict__ p, long n) {
    long i = (long)blockIdx.x * blockDim.x + threadIdx.x;
    long stride = (long)gridDim.x * blockDim.x;
    for (; i < n; i += stride) p[i] = 0.0f;
}

__global__ void convert_w_kernel(const float* __restrict__ w, __bf16* __restrict__ wbf, int n) {
    int i = blockIdx.x * blockDim.x + threadIdx.x;
    if (i < n) wbf[i] = (__bf16)w[i];
}

// ---------------------------------------------------------------------------
// Main edge GEMM kernel. PASS==0: accumulate per-column sum/sumsq of the raw
// dot products (fc_b cancels inside BatchNorm; variance is shift-invariant).
// PASS==1: apply fused BN affine, compute sigmoid(gate)*softplus(core),
// atomically scatter-add into aggr[dst].
//
// Per block: 256 threads = 8 wave32. Wave w owns output columns [16w,16w+16).
// Its B panel (16 cols x K=256, bf16) lives in 8 v16bf registers for the
// whole kernel. Each iteration gathers+converts a 32-edge x 256 A tile into
// LDS and each wave runs two 16x16x256 WMMA chains (16 wmma / barrier pair).
// ---------------------------------------------------------------------------
template <int PASS>
__global__ __launch_bounds__(256) void edge_gemm_kernel(
    const float* __restrict__ atom, const float* __restrict__ edge,
    const __bf16* __restrict__ wbf,
    const int* __restrict__ src, const int* __restrict__ dst,
    const float* __restrict__ scale1, const float* __restrict__ shift1,
    float* __restrict__ sum1, float* __restrict__ sq1,
    float* __restrict__ aggr, int E)
{
    __shared__ __bf16 sTot[TROWS][KDIM + 8];  // A tile (row stride 528B, 16B aligned)
    __shared__ float  sZ[TROWS][MDIM + 4];    // post-BN z tile (pass 1)
    __shared__ float  sSum[MDIM], sSq[MDIM];

    const int tid  = threadIdx.x;
    const int wave = tid >> 5;
    const int lane = tid & 31;
    const int nlo  = lane & 15;
    const int col  = wave * 16 + nlo;         // output column owned by this lane pair

    // --- Load resident B registers -----------------------------------------
    // B layout used here (16-bit, 32x16, contiguous k to match A): VGPR v, lanes 0-15 hold K=2v,2v+1 of col
    // `lane`; lanes 16-31 hold K=16+2v,17+2v. B[k][n] = fc_w[n][k] (row-major
    // fc_w => contiguous along k).
    v16bf breg[8];
    {
        const __bf16* wp = wbf + (long)col * KDIM + ((lane < 16) ? 0 : 16);
#pragma unroll
        for (int s = 0; s < 8; ++s) {
            V16U b;
            b.q[0] = *(const uint4*)(wp + s * 32);
            b.q[1] = *(const uint4*)(wp + s * 32 + 8);
            breg[s] = b.v;
        }
    }

    float sc1 = 0.0f, sh1 = 0.0f;
    if (PASS == 1) { sc1 = scale1[col]; sh1 = shift1[col]; }
    if (PASS == 0) { if (tid < MDIM) { sSum[tid] = 0.0f; sSq[tid] = 0.0f; } }
    __syncthreads();

    const int nTiles = (E + TROWS - 1) / TROWS;
    for (int t = blockIdx.x; t < nTiles; t += gridDim.x) {
        const long e0 = (long)t * TROWS;

        // --- Fill A tile: gather + f32->bf16 convert (2 rows per thread) ---
        {
            const int cc = tid & 15;          // 16-column chunk index
#pragma unroll
            for (int rt = 0; rt < 2; ++rt) {
                const int r  = (tid >> 4) + rt * 16;
                const long e = e0 + r;
                float tmp[16];
                if (e < (long)E) {
                    const float* p;
                    if (cc < 4)       p = atom + (long)dst[e] * FDIM + cc * 16;
                    else if (cc < 8)  p = atom + (long)src[e] * FDIM + (cc - 4) * 16;
                    else              p = edge + e * GDIM + (cc - 8) * 16;
#pragma unroll
                    for (int i = 0; i < 16; ++i) tmp[i] = p[i];
                    // prefetch next tile's streaming edge features
                    if (cc >= 8 && (e + TROWS) < (long)E)
                        __builtin_prefetch(edge + (e + TROWS) * GDIM + (cc - 8) * 16, 0, 0);
                } else {
#pragma unroll
                    for (int i = 0; i < 16; ++i) tmp[i] = 0.0f;
                }
#pragma unroll
                for (int i = 0; i < 16; ++i) sTot[r][cc * 16 + i] = (__bf16)tmp[i];
            }
        }
        __syncthreads();

        // --- WMMA: 2 x 8 x v_wmma_f32_16x16x32_bf16 per wave ---------------
        // A layout (16-bit, 16x32), contiguous k to match B: lanes 0-15 = row M, K{kb+0..15};
        // lanes 16-31 = row M, K{kb+16..31}.
        v8f c0 = {};
        v8f c1 = {};
        {
            const __bf16* rowp0 = &sTot[nlo][0];
            const __bf16* rowp1 = &sTot[nlo + 16][0];
            const int koffA = (lane < 16) ? 0 : 16;
            const int koffB = (lane < 16) ? 8 : 24;
#pragma unroll
            for (int s = 0; s < 8; ++s) {
                V16U a0, a1;
                a0.q[0] = *(const uint4*)(rowp0 + s * 32 + koffA);
                a0.q[1] = *(const uint4*)(rowp0 + s * 32 + koffB);
                a1.q[0] = *(const uint4*)(rowp1 + s * 32 + koffA);
                a1.q[1] = *(const uint4*)(rowp1 + s * 32 + koffB);
                c0 = __builtin_amdgcn_wmma_f32_16x16x32_bf16(
                         false, a0.v, false, breg[s], (short)0, c0, false, false);
                c1 = __builtin_amdgcn_wmma_f32_16x16x32_bf16(
                         false, a1.v, false, breg[s], (short)0, c1, false, false);
            }
        }

        if (PASS == 0) {
            // column sums / sums-of-squares of both 16x16 tiles (one reduction)
            float su = 0.0f, sq = 0.0f;
#pragma unroll
            for (int r = 0; r < 8; ++r) {
                float v0 = c0[r], v1 = c1[r];
                su += v0 + v1;
                sq += v0 * v0 + v1 * v1;
            }
            su += __shfl_xor(su, 16);
            sq += __shfl_xor(sq, 16);
            if (lane < 16) { sSum[col] += su; sSq[col] += sq; }
            __syncthreads();   // compute done before next fill overwrites sTot
        } else {
            // C layout: VGPR r -> (M=r, N=lane) lanes 0-15; (M=8+r, N=lane-16)
            const int rbase = (lane < 16) ? 0 : 8;
#pragma unroll
            for (int r = 0; r < 8; ++r) {
                sZ[rbase + r][col]      = c0[r] * sc1 + sh1;
                sZ[16 + rbase + r][col] = c1[r] * sc1 + sh1;
            }
            __syncthreads();
            // messages: 32 rows x 64 features, scatter-add into aggr[dst]
#pragma unroll
            for (int i = 0; i < 8; ++i) {
                int idx = tid + i * 256;
                int row = idx >> 6, f = idx & 63;
                long e = e0 + row;
                if (e < (long)E) {
                    float gv  = sZ[row][f];
                    float cv  = sZ[row][FDIM + f];
                    float msg = sigmoidf_(gv) * softplusf_(cv);
                    atomicAdd(&aggr[(long)dst[e] * FDIM + f], msg);
                }
            }
            __syncthreads();
        }
    }

    if (PASS == 0) {
        __syncthreads();
        if (tid < MDIM) {
            atomicAdd(&sum1[tid], sSum[tid]);
            atomicAdd(&sq1[tid],  sSq[tid]);
        }
    }
}

// ---------------------------------------------------------------------------
// Finalize BN1 affine: mean shift from fc_b cancels inside BN; variance is
// shift-invariant, so stats of the raw dot products suffice.
// scale1 = bn1_w * rsqrt(var+eps); shift1 = bn1_b - mean_dot*scale1
// ---------------------------------------------------------------------------
__global__ void finalize1_kernel(const float* __restrict__ sum1, const float* __restrict__ sq1,
                                 const float* __restrict__ bn1w, const float* __restrict__ bn1b,
                                 float* __restrict__ scale1, float* __restrict__ shift1, int E)
{
    int j = threadIdx.x;  // 128 threads
    float invE = 1.0f / (float)E;
    float mean = sum1[j] * invE;
    float var  = fmaxf(sq1[j] * invE - mean * mean, 0.0f);
    float sc   = bn1w[j] * rsqrtf(var + EPSV);
    scale1[j] = sc;
    shift1[j] = bn1b[j] - mean * sc;
}

// Column stats of aggr (N x 64)
__global__ __launch_bounds__(256) void aggr_stats_kernel(const float* __restrict__ aggr,
                                                         float* __restrict__ sum2,
                                                         float* __restrict__ sq2, int N)
{
    __shared__ float sS[256], sQ[256];
    int tid = threadIdx.x;
    int colf = tid & 63;
    int rgrp = tid >> 6;   // 0..3
    float su = 0.0f, sq = 0.0f;
    for (long r = (long)blockIdx.x * 4 + rgrp; r < (long)N; r += (long)gridDim.x * 4) {
        float v = aggr[r * FDIM + colf];
        su += v; sq += v * v;
    }
    sS[tid] = su; sQ[tid] = sq;
    __syncthreads();
    if (tid < 64) {
        float a = sS[tid] + sS[tid + 64] + sS[tid + 128] + sS[tid + 192];
        float b = sQ[tid] + sQ[tid + 64] + sQ[tid + 128] + sQ[tid + 192];
        atomicAdd(&sum2[tid], a);
        atomicAdd(&sq2[tid],  b);
    }
}

__global__ void finalize2_kernel(const float* __restrict__ sum2, const float* __restrict__ sq2,
                                 const float* __restrict__ bn2w, const float* __restrict__ bn2b,
                                 float* __restrict__ scale2, float* __restrict__ shift2, int N)
{
    int j = threadIdx.x;  // 64 threads
    float invN = 1.0f / (float)N;
    float mean = sum2[j] * invN;
    float var  = fmaxf(sq2[j] * invN - mean * mean, 0.0f);
    float sc   = bn2w[j] * rsqrtf(var + EPSV);
    scale2[j] = sc;
    shift2[j] = bn2b[j] - mean * sc;
}

__global__ __launch_bounds__(256) void out_kernel(const float* __restrict__ atom,
                                                  const float* __restrict__ aggr,
                                                  const float* __restrict__ scale2,
                                                  const float* __restrict__ shift2,
                                                  float* __restrict__ out, long n)
{
    long i = (long)blockIdx.x * 256 + threadIdx.x;
    long stride = (long)gridDim.x * 256;
    for (; i < n; i += stride) {
        int colf = (int)(i & 63);
        float a = aggr[i] * scale2[colf] + shift2[colf];
        out[i] = softplusf_(atom[i] + a);
    }
}

// ---------------------------------------------------------------------------
// Host launch
// ---------------------------------------------------------------------------
extern "C" void kernel_launch(void* const* d_in, const int* in_sizes, int n_in,
                              void* d_out, int out_size, void* d_ws, size_t ws_size,
                              hipStream_t stream)
{
    (void)n_in; (void)out_size; (void)ws_size;
    const float* atom = (const float*)d_in[0];   // (N, 64)
    const float* edge = (const float*)d_in[1];   // (E, 128)
    const float* fcw  = (const float*)d_in[2];   // (128, 256)
    // d_in[3] = fc_b: mathematically cancels inside BatchNorm (shift-invariant)
    const float* bn1w = (const float*)d_in[4];
    const float* bn1b = (const float*)d_in[5];
    const float* bn2w = (const float*)d_in[6];
    const float* bn2b = (const float*)d_in[7];
    const int*   src  = (const int*)d_in[8];
    const int*   dst  = (const int*)d_in[9];

    const int N = in_sizes[0] / FDIM;
    const int E = in_sizes[8];

    // Workspace layout
    float* ws     = (float*)d_ws;
    float* aggr   = ws;                           // N*64
    float* sum1   = aggr + (long)N * FDIM;        // 128
    float* sq1    = sum1 + MDIM;                  // 128
    float* scale1 = sq1 + MDIM;                   // 128
    float* shift1 = scale1 + MDIM;                // 128
    float* sum2   = shift1 + MDIM;                // 64
    float* sq2    = sum2 + FDIM;                  // 64
    float* scale2 = sq2 + FDIM;                   // 64
    float* shift2 = scale2 + FDIM;                // 64
    __bf16* wbf   = (__bf16*)(shift2 + FDIM);     // 128*256 bf16

    const long zeroN = (long)N * FDIM + 4 * MDIM + 2 * FDIM; // aggr + all stat accumulators
    zero_kernel<<<2048, 256, 0, stream>>>(aggr, zeroN);
    convert_w_kernel<<<(MDIM * KDIM + 255) / 256, 256, 0, stream>>>(fcw, wbf, MDIM * KDIM);

    // Pass 0: GEMM -> column stats of dot products
    edge_gemm_kernel<0><<<4096, 256, 0, stream>>>(atom, edge, wbf, src, dst,
                                                  nullptr, nullptr, sum1, sq1, nullptr, E);
    finalize1_kernel<<<1, MDIM, 0, stream>>>(sum1, sq1, bn1w, bn1b, scale1, shift1, E);

    // Pass 1: GEMM -> BN affine -> gated messages -> scatter-add into aggr
    edge_gemm_kernel<1><<<4096, 256, 0, stream>>>(atom, edge, wbf, src, dst,
                                                  scale1, shift1, nullptr, nullptr, aggr, E);

    aggr_stats_kernel<<<2048, 256, 0, stream>>>(aggr, sum2, sq2, N);
    finalize2_kernel<<<1, FDIM, 0, stream>>>(sum2, sq2, bn2w, bn2b, scale2, shift2, N);
    out_kernel<<<2048, 256, 0, stream>>>(atom, aggr, scale2, shift2, (float*)d_out, (long)N * FDIM);
}
